// ODCRN_5145370820941
// MI455X (gfx1250) — compile-verified, matrix operand order checked by reading, not run on hardware
//
#include <hip/hip_runtime.h>
#include <math.h>

typedef __bf16 bf16;
typedef __attribute__((ext_vector_type(16))) __bf16 v16bf;
typedef __attribute__((ext_vector_type(8)))  __bf16 v8bf;
typedef __attribute__((ext_vector_type(4)))  __bf16 v4bf;
typedef __attribute__((ext_vector_type(8)))  float  v8f;

#define NN  512
#define BB  8
#define TT  12
#define HH  64

// ---------------------------------------------------------------------------
// Workhorse: batched bf16 GEMM, C = A(MxK) * B(KxN) (+bias), fp32 accumulate
// via v_wmma_f32_16x16x32_bf16.  Requirements (all call sites satisfy):
//   M % 128 == 0, N % 32 == 0, K % 32 == 0, lda % 8 == 0, ldb % 4 == 0.
// Block: 256 threads (8 waves). Each block computes a 128x32 C tile; each
// wave owns a 16x32 slice (two 16x16 WMMA accumulators).
// B tile is stored transposed in LDS so both operands are fetched with
// 16-byte ds_load_b128; tiles are double-buffered to overlap global loads
// with the matrix pipe.
// ---------------------------------------------------------------------------
__global__ __launch_bounds__(256)
void gemm_bf16_wmma(const bf16* __restrict__ A, long long sA, int lda,
                    const bf16* __restrict__ Bm, long long sB, int ldb,
                    const float* __restrict__ bias,
                    void* __restrict__ Cout, long long sC, int ldc, int c_bf16,
                    int K)
{
    __shared__ __align__(16) bf16 As[2][128 * 32];  // 2 x 8 KB
    __shared__ __align__(16) bf16 Bt[2][32 * 40];   // transposed, stride 40 halves

    const int tid  = threadIdx.x;
    const int wave = tid >> 5;
    const int lane = tid & 31;
    const int lrow = lane & 15;
    const int kgrp = (lane >> 4) << 3;              // 0 or 8 (K sub-group)

    const int m0 = blockIdx.y * 128;
    const int n0 = blockIdx.x * 32;

    A  += (long long)blockIdx.z * sA;
    Bm += (long long)blockIdx.z * sB;

    v8f acc0 = {0.f,0.f,0.f,0.f,0.f,0.f,0.f,0.f};
    v8f acc1 = {0.f,0.f,0.f,0.f,0.f,0.f,0.f,0.f};

    // per-thread tile-load coordinates
    const int ar0 = tid >> 2;                       // A rows: pass0 -> 0..63
    const int ac  = (tid & 3) << 3;                 // A col segment (8 halves)
    const int br  = tid >> 3;                       // B row (k) 0..31
    const int bc  = (tid & 7) << 2;                 // B col segment (4 halves)

    auto load_tile = [&](int kk, int buf) {
        // A tile: 128 rows x 32 halves, 16-byte vector copies
        *(v8bf*)&As[buf][ar0 * 32 + ac] =
            *(const v8bf*)&A[(long long)(m0 + ar0) * lda + kk + ac];
        *(v8bf*)&As[buf][(ar0 + 64) * 32 + ac] =
            *(const v8bf*)&A[(long long)(m0 + ar0 + 64) * lda + kk + ac];
        // B tile: 32x32, loaded row-wise (8B vector), stored transposed
        v4bf bv = *(const v4bf*)&Bm[(long long)(kk + br) * ldb + n0 + bc];
        Bt[buf][(bc + 0) * 40 + br] = bv[0];
        Bt[buf][(bc + 1) * 40 + br] = bv[1];
        Bt[buf][(bc + 2) * 40 + br] = bv[2];
        Bt[buf][(bc + 3) * 40 + br] = bv[3];
    };

    load_tile(0, 0);
    __syncthreads();

    int cur = 0;
    const int arow = (wave * 16 + lrow) * 32;
    for (int kk = 0; kk < K; kk += 32) {
        int nxt = cur ^ 1;
        if (kk + 32 < K) {
            load_tile(kk + 32, nxt);
            if (kk + 64 < K)    // hint the tile after next (global_prefetch_b8)
                __builtin_prefetch(&A[(long long)(m0 + (tid >> 1)) * lda + kk + 64], 0, 1);
        }

        v8bf alo = *(const v8bf*)&As[cur][arow + kgrp];
        v8bf ahi = *(const v8bf*)&As[cur][arow + kgrp + 16];
        v16bf av = __builtin_shufflevector(alo, ahi,
                    0,1,2,3,4,5,6,7,8,9,10,11,12,13,14,15);
        v8bf b0lo = *(const v8bf*)&Bt[cur][lrow * 40 + kgrp];
        v8bf b0hi = *(const v8bf*)&Bt[cur][lrow * 40 + kgrp + 16];
        v16bf b0 = __builtin_shufflevector(b0lo, b0hi,
                    0,1,2,3,4,5,6,7,8,9,10,11,12,13,14,15);
        v8bf b1lo = *(const v8bf*)&Bt[cur][(16 + lrow) * 40 + kgrp];
        v8bf b1hi = *(const v8bf*)&Bt[cur][(16 + lrow) * 40 + kgrp + 16];
        v16bf b1 = __builtin_shufflevector(b1lo, b1hi,
                    0,1,2,3,4,5,6,7,8,9,10,11,12,13,14,15);

        acc0 = __builtin_amdgcn_wmma_f32_16x16x32_bf16(false, av, false, b0,
                                                       (short)0, acc0, false, false);
        acc1 = __builtin_amdgcn_wmma_f32_16x16x32_bf16(false, av, false, b1,
                                                       (short)0, acc1, false, false);
        __syncthreads();
        cur = nxt;
    }

    float bia0 = bias ? bias[n0 + lrow]      : 0.f;
    float bia1 = bias ? bias[n0 + 16 + lrow] : 0.f;
    int baseM  = m0 + wave * 16 + (lane >> 4) * 8;  // C layout: VGPR g -> M = base+g

    if (c_bf16) {
        bf16* C = (bf16*)Cout + (long long)blockIdx.z * sC;
        #pragma unroll
        for (int g = 0; g < 8; ++g) {
            long long ro = (long long)(baseM + g) * ldc;
            C[ro + n0 + lrow]      = (bf16)(acc0[g] + bia0);
            C[ro + n0 + 16 + lrow] = (bf16)(acc1[g] + bia1);
        }
    } else {
        float* C = (float*)Cout + (long long)blockIdx.z * sC;
        #pragma unroll
        for (int g = 0; g < 8; ++g) {
            long long ro = (long long)(baseM + g) * ldc;
            C[ro + n0 + lrow]      = acc0[g] + bia0;
            C[ro + n0 + 16 + lrow] = acc1[g] + bia1;
        }
    }
}

// ---------------------------------------------------------------------------
// Support-matrix preparation for one timestep t
// ---------------------------------------------------------------------------
__global__ __launch_bounds__(128)
void rownorm_k(const float* __restrict__ od, bf16* __restrict__ Af, int t)
{
    int row = blockIdx.x;                 // b*512 + m
    int b = row >> 9, m = row & 511;
    const float* src = od + ((((long long)b * TT + t) * NN + m) << 9);
    int tid = threadIdx.x;
    float s = 0.f;
    #pragma unroll
    for (int i = 0; i < 4; ++i) s += src[tid + i * 128];
    __shared__ float red[128];
    red[tid] = s; __syncthreads();
    for (int o = 64; o > 0; o >>= 1) {
        if (tid < o) red[tid] += red[tid + o];
        __syncthreads();
    }
    float inv = 1.f / (red[0] + 1e-8f);
    bf16* dst = Af + (((long long)b * NN + m) << 9);
    #pragma unroll
    for (int i = 0; i < 4; ++i) dst[tid + i * 128] = (bf16)(src[tid + i * 128] * inv);
}

__global__ void colsum_k(const float* __restrict__ od, float* __restrict__ cs, int t)
{
    int b = blockIdx.y;
    int m = blockIdx.x * 256 + threadIdx.x;
    const float* base = od + (((long long)b * TT + t) * NN) * NN;
    float s = 0.f;
    for (int n = 0; n < NN; ++n) s += base[(long long)n * NN + m];
    cs[b * NN + m] = s;
}

__global__ __launch_bounds__(256)
void transnorm_k(const float* __restrict__ od, const float* __restrict__ cs,
                 bf16* __restrict__ Ab, int t)
{
    __shared__ float tile[32][33];
    int b  = blockIdx.z;
    int n0 = blockIdx.x * 32, m0 = blockIdx.y * 32;
    int tx = threadIdx.x & 31, ty = threadIdx.x >> 5;   // 32 x 8
    const float* base = od + (((long long)b * TT + t) * NN) * NN;
    #pragma unroll
    for (int q = 0; q < 4; ++q)
        tile[ty + q * 8][tx] = base[(long long)(n0 + ty + q * 8) * NN + m0 + tx];
    __syncthreads();
    #pragma unroll
    for (int q = 0; q < 4; ++q) {
        int m = m0 + ty + q * 8;
        float inv = 1.f / (cs[b * NN + m] + 1e-8f);
        Ab[(((long long)b * NN + m) << 9) + n0 + tx] = (bf16)(tile[tx][ty + q * 8] * inv);
    }
}

// ---------------------------------------------------------------------------
// Small helpers (sub-1% FLOPs)
// ---------------------------------------------------------------------------
__global__ void matvec_k(const float* __restrict__ Amat, const float* __restrict__ x,
                         int xstride, float* __restrict__ y)
{
    int idx = blockIdx.x * 256 + threadIdx.x;   // BT*512
    int bt = idx >> 9, m = idx & 511;
    const float* xr = x + (long long)bt * NN * xstride;
    const float* ar = Amat + (long long)m * NN;
    float s = 0.f;
    for (int n = 0; n < NN; ++n) s += ar[n] * xr[n * xstride];
    y[idx] = s;
}

__global__ void gconv0_k(const float* __restrict__ crowd, int ch,
                         const float* __restrict__ s1, const float* __restrict__ s2,
                         const float* __restrict__ W, const float* __restrict__ bvec,
                         float* __restrict__ out)
{
    int idx = blockIdx.x * 256 + threadIdx.x;   // BT*512*32
    int row = idx >> 5, j = idx & 31;
    float x0 = crowd[(long long)row * 2 + ch];
    out[idx] = x0 * W[j] + s1[row] * W[32 + j] + s2[row] * W[64 + j] + bvec[j];
}

__global__ void naive_gemm_k(const float* __restrict__ A, const float* __restrict__ Bm,
                             const float* __restrict__ bias, float* __restrict__ C,
                             int M, int Nc, int K, int relu)
{
    int idx = blockIdx.x * 256 + threadIdx.x;
    if (idx >= M * Nc) return;
    int m = idx / Nc, j = idx - m * Nc;
    float s = bias ? bias[j] : 0.f;
    for (int k = 0; k < K; ++k) s += A[m * K + k] * Bm[k * Nc + j];
    if (relu && s < 0.f) s = 0.f;
    C[idx] = s;
}

__global__ void build_x_k(const float* __restrict__ srch, const float* __restrict__ dsth,
                          const float* __restrict__ e, const float* __restrict__ state,
                          const float* __restrict__ zr,   // null -> gate input, else r*state
                          bf16* __restrict__ feats, int t)
{
    int idx = blockIdx.x * 256 + threadIdx.x;   // B*512*160
    int row = idx / 160, f = idx - row * 160;
    int b = row >> 9, n = row & 511;
    long long sb = (((long long)b * TT + t) * NN + n) * 32;
    float v;
    if      (f < 32) v = srch[sb + f];
    else if (f < 64) v = dsth[sb + f - 32];
    else if (f < 96) v = e[n * 32 + f - 64];
    else {
        float st = state[(long long)row * 64 + f - 96];
        if (zr) { float r = zr[(long long)row * 128 + 64 + (f - 96)]; v = r * st; }
        else v = st;
    }
    feats[(long long)row * 800 + f] = (bf16)v;
}

__global__ void sigmoid_k(float* __restrict__ x, int n)
{
    int idx = blockIdx.x * 256 + threadIdx.x;
    if (idx < n) x[idx] = 1.f / (1.f + expf(-x[idx]));
}

__global__ void update_k(float* __restrict__ state, const float* __restrict__ zr,
                         const float* __restrict__ cc)
{
    int idx = blockIdx.x * 256 + threadIdx.x;   // B*512*64
    int row = idx >> 6, h = idx & 63;
    float z = zr[(long long)row * 128 + h];
    float c = tanhf(cc[idx]);
    state[idx] = z * state[idx] + (1.f - z) * c;
}

__global__ void zero_k(float* __restrict__ p, int n)
{
    int idx = blockIdx.x * 256 + threadIdx.x;
    if (idx < n) p[idx] = 0.f;
}

__global__ void cvt_k(const float* __restrict__ s, bf16* __restrict__ d, int n)
{
    int idx = blockIdx.x * 256 + threadIdx.x;
    if (idx < n) d[idx] = (bf16)s[idx];
}

__global__ void cvt_ld_k(const float* __restrict__ s, bf16* __restrict__ d,
                         int rows, int cols, int ldd)
{
    int idx = blockIdx.x * 256 + threadIdx.x;
    if (idx >= rows * cols) return;
    int r = idx / cols, c = idx - r * cols;
    d[(long long)r * ldd + c] = (bf16)s[idx];
}

// ---------------------------------------------------------------------------
extern "C" void kernel_launch(void* const* d_in, const int* in_sizes, int n_in,
                              void* d_out, int out_size, void* d_ws, size_t ws_size,
                              hipStream_t stream)
{
    (void)in_sizes; (void)n_in; (void)out_size; (void)ws_size;
    const float* od    = (const float*)d_in[0];
    const float* adj   = (const float*)d_in[1];
    const float* crowd = (const float*)d_in[2];
    const float* emb   = (const float*)d_in[3];
    const float* P[26];
    for (int i = 0; i < 26; ++i) P[i] = (const float*)d_in[4 + i];
    const float *gc_src_W[2] = {P[0], P[12]}, *gc_src_b[2] = {P[1], P[13]};
    const float *gc_dst_W[2] = {P[2], P[14]}, *gc_dst_b[2] = {P[3], P[15]};
    const float *mlp_W0[2]   = {P[4], P[16]}, *mlp_b0[2]   = {P[5], P[17]};
    const float *mlp_W1[2]   = {P[6], P[18]}, *mlp_b1[2]   = {P[7], P[19]};
    const float *Wg[2] = {P[8], P[20]}, *bg[2] = {P[9],  P[21]};
    const float *Wc[2] = {P[10],P[22]}, *bc[2] = {P[11], P[23]};
    const float *decW = P[24], *decb = P[25];

    char* wsb = (char*)d_ws;
    size_t off = 0;
    auto alloc = [&](size_t bytes) -> char* {
        char* p = wsb + off;
        off = (off + bytes + 255) & ~(size_t)255;
        return p;
    };
    const int BT = BB * TT;
    bf16*  Af     = (bf16*) alloc((size_t)BB * NN * NN * 2);
    bf16*  Ab     = (bf16*) alloc((size_t)BB * NN * NN * 2);
    float* cs     = (float*)alloc((size_t)BB * NN * 4);
    bf16*  adjb   = (bf16*) alloc((size_t)NN * NN * 2);
    float* srch0  = (float*)alloc((size_t)BT * NN * 32 * 4);
    float* dsth0  = (float*)alloc((size_t)BT * NN * 32 * 4);
    float* srch1  = (float*)alloc((size_t)BT * NN * 32 * 4);
    float* dsth1  = (float*)alloc((size_t)BT * NN * 32 * 4);
    bf16*  x96    = (bf16*) alloc((size_t)BT * NN * 96 * 2);
    float* s1     = (float*)alloc((size_t)BT * NN * 4);
    float* s2     = (float*)alloc((size_t)BT * NN * 4);
    float* hmlp   = (float*)alloc((size_t)NN * 32 * 4);
    float* evec   = (float*)alloc((size_t)NN * 32 * 4);
    float* state  = (float*)alloc((size_t)BB * NN * HH * 4);
    bf16*  fg     = (bf16*) alloc((size_t)BB * NN * 800 * 2);
    bf16*  fc     = (bf16*) alloc((size_t)BB * NN * 800 * 2);
    float* zr     = (float*)alloc((size_t)BB * NN * 128 * 4);
    float* cc     = (float*)alloc((size_t)BB * NN * 64 * 4);
    bf16*  WgB[2] = {(bf16*)alloc(800 * 128 * 2), (bf16*)alloc(800 * 128 * 2)};
    bf16*  WcB[2] = {(bf16*)alloc(800 * 64 * 2),  (bf16*)alloc(800 * 64 * 2)};
    bf16*  gsWb   = (bf16*) alloc(96 * 32 * 2);
    bf16*  gdWb   = (bf16*) alloc(96 * 32 * 2);
    bf16*  decWb  = (bf16*) alloc(64 * 512 * 2);
    bf16*  stateB = (bf16*) alloc((size_t)BB * NN * HH * 2);

    // ---- one-time weight conversions to bf16 ----
    cvt_k<<<(NN*NN + 255)/256, 256, 0, stream>>>(adj, adjb, NN*NN);
    cvt_k<<<(800*128 + 255)/256, 256, 0, stream>>>(Wg[0], WgB[0], 800*128);
    cvt_k<<<(800*128 + 255)/256, 256, 0, stream>>>(Wg[1], WgB[1], 800*128);
    cvt_k<<<(800*64  + 255)/256, 256, 0, stream>>>(Wc[0], WcB[0], 800*64);
    cvt_k<<<(800*64  + 255)/256, 256, 0, stream>>>(Wc[1], WcB[1], 800*64);
    cvt_k<<<(96*32   + 255)/256, 256, 0, stream>>>(gc_src_W[1], gsWb, 96*32);
    cvt_k<<<(96*32   + 255)/256, 256, 0, stream>>>(gc_dst_W[1], gdWb, 96*32);
    cvt_k<<<(64*512  + 255)/256, 256, 0, stream>>>(decW, decWb, 64*512);

    // ---- layer-0 graph conv (din=1): src = crowd[...,0], dst = crowd[...,1] ----
    matvec_k<<<BT*NN/256, 256, 0, stream>>>(adj, crowd + 0, 2, s1);
    matvec_k<<<BT*NN/256, 256, 0, stream>>>(adj, s1, 1, s2);
    gconv0_k<<<BT*NN*32/256, 256, 0, stream>>>(crowd, 0, s1, s2, gc_src_W[0], gc_src_b[0], srch0);
    matvec_k<<<BT*NN/256, 256, 0, stream>>>(adj, crowd + 1, 2, s1);
    matvec_k<<<BT*NN/256, 256, 0, stream>>>(adj, s1, 1, s2);
    gconv0_k<<<BT*NN*32/256, 256, 0, stream>>>(crowd, 1, s1, s2, gc_dst_W[0], gc_dst_b[0], dsth0);

    const long long sAf = (long long)NN * NN;      // bf16 elements per batch matrix
    const long long sF  = (long long)NN * 800;
    const long long s96 = (long long)NN * 96;

    for (int l = 0; l < 2; ++l) {
        const float* sh = (l == 0) ? srch0 : srch1;
        const float* dh = (l == 0) ? dsth0 : dsth1;

        if (l == 1) {
            // layer-1 graph conv (din=32) via WMMA: [x, adj@x, adj@adj@x] @ W + b
            const float* inp[2] = {srch0, dsth0};
            const bf16*  Wb2[2] = {gsWb,  gdWb};
            const float* bb2[2] = {gc_src_b[1], gc_dst_b[1]};
            float*       out2[2]= {srch1, dsth1};
            for (int s = 0; s < 2; ++s) {
                cvt_ld_k<<<BT*NN*32/256, 256, 0, stream>>>(inp[s], x96, BT*NN, 32, 96);
                dim3 g1(1, 4, BT);
                gemm_bf16_wmma<<<g1, 256, 0, stream>>>(adjb, 0, NN, x96 + 0,  s96, 96,
                    nullptr, x96 + 32, s96, 96, 1, NN);
                gemm_bf16_wmma<<<g1, 256, 0, stream>>>(adjb, 0, NN, x96 + 32, s96, 96,
                    nullptr, x96 + 64, s96, 96, 1, NN);
                gemm_bf16_wmma<<<g1, 256, 0, stream>>>(x96, s96, 96, Wb2[s], 0, 32,
                    bb2[s], out2[s], (long long)NN*32, 32, 0, 96);
            }
        }

        // context embedding e = relu(emb@W0 + b0)@W1 + b1   (shared across batch)
        naive_gemm_k<<<(NN*32 + 255)/256, 256, 0, stream>>>(emb,  mlp_W0[l], mlp_b0[l], hmlp, NN, 32, 16, 1);
        naive_gemm_k<<<(NN*32 + 255)/256, 256, 0, stream>>>(hmlp, mlp_W1[l], mlp_b1[l], evec, NN, 32, 32, 0);

        zero_k<<<(BB*NN*HH + 255)/256, 256, 0, stream>>>(state, BB*NN*HH);

        for (int t = 0; t < TT; ++t) {
            // supports for this timestep (bf16)
            rownorm_k  <<<BB*NN, 128, 0, stream>>>(od, Af, t);
            colsum_k   <<<dim3(2, BB), 256, 0, stream>>>(od, cs, t);
            transnorm_k<<<dim3(16, 16, BB), 256, 0, stream>>>(od, cs, Ab, t);

            // ---- gate path ----
            build_x_k<<<BB*NN*160/256, 256, 0, stream>>>(sh, dh, evec, state, nullptr, fg, t);
            dim3 gd(5, 4, BB);
            gemm_bf16_wmma<<<gd, 256, 0, stream>>>(Af, sAf, NN, fg + 0,   sF, 800, nullptr, fg + 160, sF, 800, 1, NN);
            gemm_bf16_wmma<<<gd, 256, 0, stream>>>(Af, sAf, NN, fg + 160, sF, 800, nullptr, fg + 320, sF, 800, 1, NN);
            gemm_bf16_wmma<<<gd, 256, 0, stream>>>(Ab, sAf, NN, fg + 0,   sF, 800, nullptr, fg + 480, sF, 800, 1, NN);
            gemm_bf16_wmma<<<gd, 256, 0, stream>>>(Ab, sAf, NN, fg + 480, sF, 800, nullptr, fg + 640, sF, 800, 1, NN);
            gemm_bf16_wmma<<<dim3(4, 4, BB), 256, 0, stream>>>(fg, sF, 800, WgB[l], 0, 128,
                bg[l], zr, (long long)NN*128, 128, 0, 800);
            sigmoid_k<<<(BB*NN*128 + 255)/256, 256, 0, stream>>>(zr, BB*NN*128);

            // ---- candidate path ----
            build_x_k<<<BB*NN*160/256, 256, 0, stream>>>(sh, dh, evec, state, zr, fc, t);
            gemm_bf16_wmma<<<gd, 256, 0, stream>>>(Af, sAf, NN, fc + 0,   sF, 800, nullptr, fc + 160, sF, 800, 1, NN);
            gemm_bf16_wmma<<<gd, 256, 0, stream>>>(Af, sAf, NN, fc + 160, sF, 800, nullptr, fc + 320, sF, 800, 1, NN);
            gemm_bf16_wmma<<<gd, 256, 0, stream>>>(Ab, sAf, NN, fc + 0,   sF, 800, nullptr, fc + 480, sF, 800, 1, NN);
            gemm_bf16_wmma<<<gd, 256, 0, stream>>>(Ab, sAf, NN, fc + 480, sF, 800, nullptr, fc + 640, sF, 800, 1, NN);
            gemm_bf16_wmma<<<dim3(2, 4, BB), 256, 0, stream>>>(fc, sF, 800, WcB[l], 0, 64,
                bc[l], cc, (long long)NN*64, 64, 0, 800);

            // state = z*state + (1-z)*tanh(cc)
            update_k<<<(BB*NN*HH + 255)/256, 256, 0, stream>>>(state, zr, cc);
        }
    }

    // ---- decoder: out = state @ dec_W + dec_b  -> (B, 512, 512) fp32 ----
    cvt_k<<<(BB*NN*HH + 255)/256, 256, 0, stream>>>(state, stateB, BB*NN*HH);
    gemm_bf16_wmma<<<dim3(16, 4, BB), 256, 0, stream>>>(stateB, (long long)NN*HH, HH,
        decWb, 0, NN, decb, (float*)d_out, (long long)NN*NN, NN, 0, HH);
}
